// self_attention_22823456211579
// MI455X (gfx1250) — matrix operand order repaired and checked by reading, not compiled, not run on hardware
//
#include <hip/hip_runtime.h>
#include <hip/hip_bf16.h>

// ---------------------------------------------------------------------------
// Multi-head self-attention for MI455X (gfx1250), wave32 + WMMA bf16.
//   B=2, S=2048, D=1024, H=16, Dh=64
// Pipeline:
//   1) fp32 -> bf16 converts (activations) and transposed converts (weights)
//   2) bf16 WMMA GEMM with LDS double-buffering; global->LDS staging uses
//      GLOBAL_LOAD_ASYNC_TO_LDS_B128 (ASYNCcnt) when the builtin exists.
//   3) flash-attention kernel: one wave per (b,h,16-row q tile), online
//      softmax, K-fragment prefetch, P staged through LDS for the 2nd WMMA.
// ---------------------------------------------------------------------------

typedef __attribute__((ext_vector_type(16))) __bf16 v16bf;
typedef __attribute__((ext_vector_type(8)))  __bf16 v8bf;
typedef __attribute__((ext_vector_type(8)))  float  v8f;
typedef __attribute__((ext_vector_type(4)))  float  v4f;
typedef __attribute__((ext_vector_type(4)))  int    v4i;

#define BATCH 2
#define SEQ   2048
#define DM    1024
#define NH    16
#define DH    64
#define MTOT  (BATCH * SEQ)   // 4096

#define WMMA_BF16(A, B, C) \
    __builtin_amdgcn_wmma_f32_16x16x32_bf16(false, (A), false, (B), (short)0, (C), false, false)

#define AS1 __attribute__((address_space(1)))
#define AS3 __attribute__((address_space(3)))

#if defined(__gfx1250__) && __has_builtin(__builtin_amdgcn_global_load_async_to_lds_b128)
#define HAVE_ASYNC_LDS 1
#else
#define HAVE_ASYNC_LDS 0
#endif

#if HAVE_ASYNC_LDS
#  if __has_builtin(__builtin_amdgcn_s_wait_asynccnt)
#    define WAIT_CP(n) __builtin_amdgcn_s_wait_asynccnt(n)
#  else
#    define WAIT_CP(n) asm volatile("s_wait_asynccnt %0" :: "i"(n) : "memory")
#  endif
#else
#  define WAIT_CP(n) ((void)0)
#endif

// Copy 16 bytes (8 bf16) global -> LDS for this lane (async, ASYNCcnt-tracked).
__device__ __forceinline__ void cp16_g2l(const __bf16* g, __bf16* l) {
#if HAVE_ASYNC_LDS
    __builtin_amdgcn_global_load_async_to_lds_b128(
        (AS1 v4i*)(unsigned long long)g,
        (AS3 v4i*)(unsigned int)(unsigned long long)l, 0, 0);
#else
    *reinterpret_cast<v8bf*>(l) = *reinterpret_cast<const v8bf*>(g);
#endif
}

// Load a 16x32 bf16 A-fragment row for this lane.
// Layout (contiguous K order): lane<16 holds M=lane, K in {k0..k0+15};
// lane>=16 holds M=lane-16, K in {k0+16..k0+31}.  Caller passes p already offset by
// row*ld + k0 + 16*(lane>>4); we fetch 16 contiguous elements as two 8-element runs.
__device__ __forceinline__ v16bf load_a16(const __bf16* p) {
    v8bf lo = *reinterpret_cast<const v8bf*>(p);
    v8bf hi = *reinterpret_cast<const v8bf*>(p + 8);
    v16bf r;
#pragma unroll
    for (int i = 0; i < 8; ++i) { r[i] = lo[i]; r[i + 8] = hi[i]; }
    return r;
}

// ---------------------------------------------------------------------------
// fp32 -> bf16 elementwise, 4 elements per thread (float4 in, 4x bf16 out)
__global__ __launch_bounds__(256) void cvt_bf16_kernel(const float* __restrict__ in,
                                                       __bf16* __restrict__ out, int n4) {
    int i = blockIdx.x * blockDim.x + threadIdx.x;
    if (i < n4) {
        v4f f = reinterpret_cast<const v4f*>(in)[i];
        __bf16 o[4] = {(__bf16)f.x, (__bf16)f.y, (__bf16)f.z, (__bf16)f.w};
        *reinterpret_cast<unsigned long long*>(out + 4 * (size_t)i) =
            *reinterpret_cast<unsigned long long*>(o);
    }
}

// fp32 [K][N] -> bf16 [N][K] (transpose weights so B-fragments are contiguous)
__global__ __launch_bounds__(256) void cvtT_bf16_kernel(const float* __restrict__ in,
                                                        __bf16* __restrict__ out) {
    int i = blockIdx.x * blockDim.x + threadIdx.x;
    if (i < DM * DM) {
        int k = i >> 10, n = i & (DM - 1);
        out[(size_t)n * DM + k] = (__bf16)in[i];
    }
}

// ---------------------------------------------------------------------------
// GEMM: C[4096 x 1024] = X[4096 x 1024] * W  (W given transposed: WT[N][K])
// Block = 256 threads = 8 waves (4 row-waves x 2 col-waves); block tile
// 64 rows x 256 cols; K-step 32 staged through double-buffered LDS with
// async global->LDS copies (5 x b128 per thread per step).
// mode 0: store bf16 head layout   [B,H,S,Dh]
// mode 1: store bf16 head-T layout [B,H,Dh,S]
// mode 2: store fp32 [M,N] with bias
__global__ __launch_bounds__(256)
void gemm_bf16_kernel(const __bf16* __restrict__ X, const __bf16* __restrict__ WT,
                      const float* __restrict__ bias, void* __restrict__ outp, int mode) {
    __shared__ alignas(16) __bf16 ldsA[2][64 * 32];    // 2 x 4 KB
    __shared__ alignas(16) __bf16 ldsB[2][256 * 32];   // 2 x 16 KB

    const int tid  = threadIdx.x;
    const int lane = tid & 31;
    const int wid  = tid >> 5;
    const int bcol = blockIdx.x & 3;          // 1024/256 = 4 col tiles
    const int brow = blockIdx.x >> 2;         // 4096/64  = 64 row tiles
    const int wrow = wid >> 1, wcol = wid & 1;
    const int rowBase = brow * 64;
    const int colBase = bcol * 256;
    const int half = lane >> 4, ln = lane & 15;

    // per-thread staging assignment
    const int ai = tid >> 2;                  // A-tile row 0..63
    const int ak = (tid & 3) * 8;             // A-tile k sub-chunk

    auto stage = [&](int k, int buf) {
        cp16_g2l(X + (size_t)(rowBase + ai) * DM + k + ak, &ldsA[buf][ai * 32 + ak]);
        const __bf16* g = WT + (size_t)(colBase + tid) * DM + k;
        __bf16* l = &ldsB[buf][tid * 32];
#pragma unroll
        for (int j = 0; j < 4; ++j) cp16_g2l(g + 8 * j, l + 8 * j);
    };

    v8f c[8] = {};
    stage(0, 0);
    int cur = 0;

    for (int k = 0; k < DM; k += 32) {
        if (k + 32 < DM) {
            stage(k + 32, cur ^ 1);   // 5 async copies into the other buffer
            WAIT_CP(5);               // retire this buffer's copies, next 5 in flight
        } else {
            WAIT_CP(0);
        }
        __syncthreads();              // all waves' copies visible

        v16bf a = load_a16(&ldsA[cur][(wrow * 16 + ln) * 32 + 16 * half]);
#pragma unroll
        for (int t = 0; t < 8; ++t) {
            v16bf b = *reinterpret_cast<const v16bf*>(
                &ldsB[cur][(wcol * 128 + 16 * t + ln) * 32 + 16 * half]);
            c[t] = WMMA_BF16(a, b, c[t]);
        }
        __syncthreads();              // done reading before buffer is re-filled
        cur ^= 1;
    }

#pragma unroll
    for (int t = 0; t < 8; ++t) {
#pragma unroll
        for (int r = 0; r < 8; ++r) {
            float v  = c[t][r];
            int   gr = rowBase + wrow * 16 + r + 8 * half;   // global row (b*S + s)
            int   cl = colBase + wcol * 128 + 16 * t + ln;   // global col
            if (mode == 2) {
                ((float*)outp)[(size_t)gr * DM + cl] = v + bias[cl];
            } else {
                int bb = gr >> 11, s = gr & (SEQ - 1);
                int hh = cl >> 6,  dh = cl & (DH - 1);
                size_t idx = (mode == 0)
                    ? ((size_t)((bb * NH + hh) * SEQ + s)) * DH + dh
                    : ((size_t)((bb * NH + hh) * DH + dh)) * SEQ + s;
                ((__bf16*)outp)[idx] = (__bf16)v;
            }
        }
    }
}

// ---------------------------------------------------------------------------
// Flash attention: one wave handles one (b, h, 16-row q tile); loops over the
// 2048 keys in chunks of 32.  Per chunk: 4 WMMAs for S = Q*K^T, mask + scale,
// online softmax (shfl_xor row reductions), P -> LDS -> A-fragment, 4 WMMAs
// for O += P*V.  K fragments for the next chunk are prefetched; V/mask loads
// are hoisted to the iteration top so they overlap the softmax VALU work.
__global__ __launch_bounds__(256)
void flash_attn_kernel(const __bf16* __restrict__ Qh, const __bf16* __restrict__ Kh,
                       const __bf16* __restrict__ Vt, const int* __restrict__ mask,
                       __bf16* __restrict__ out) {
    __shared__ __bf16 pbuf[8 * 16 * 32];          // 1 KB per wave
    const int lane = threadIdx.x & 31;
    const int wid  = threadIdx.x >> 5;
    const int gw   = blockIdx.x * 8 + wid;        // 4096 waves total
    const int qt   = gw & 127;                    // 128 q tiles of 16 rows
    const int h    = (gw >> 7) & (NH - 1);
    const int b    = gw >> 11;
    const int half = lane >> 4, ln = lane & 15;

    const __bf16* qbase = Qh + ((size_t)(b * NH + h) * SEQ) * DH;
    const __bf16* kbase = Kh + ((size_t)(b * NH + h) * SEQ) * DH;
    const __bf16* vbase = Vt + ((size_t)(b * NH + h) * DH) * SEQ;
    const int*    mbase = mask + (size_t)b * SEQ * SEQ;
    __bf16*       pmine = pbuf + wid * (16 * 32);

    // Q fragments for this 16-row tile (Dh=64 -> two K=32 fragments)
    v16bf aq0 = load_a16(qbase + (size_t)(qt * 16 + ln) * DH + 0  + 16 * half);
    v16bf aq1 = load_a16(qbase + (size_t)(qt * 16 + ln) * DH + 32 + 16 * half);

    auto loadK = [&](int kc, v16bf* dst) {
        dst[0] = *reinterpret_cast<const v16bf*>(kbase + (size_t)(kc + ln) * DH + 16 * half);
        dst[1] = *reinterpret_cast<const v16bf*>(kbase + (size_t)(kc + ln) * DH + 32 + 16 * half);
        dst[2] = *reinterpret_cast<const v16bf*>(kbase + (size_t)(kc + 16 + ln) * DH + 16 * half);
        dst[3] = *reinterpret_cast<const v16bf*>(kbase + (size_t)(kc + 16 + ln) * DH + 32 + 16 * half);
    };

    v8f   oc[4] = {};
    float mrow[8], lrow[8];
#pragma unroll
    for (int r = 0; r < 8; ++r) { mrow[r] = -1e30f; lrow[r] = 0.f; }

    v16bf bk[4];
    loadK(0, bk);

    for (int kc = 0; kc < SEQ; kc += 32) {
        // ---- hoisted loads: V fragments + mask words for this chunk
        v16bf bv[4];
#pragma unroll
        for (int t = 0; t < 4; ++t)
            bv[t] = *reinterpret_cast<const v16bf*>(
                vbase + (size_t)(16 * t + ln) * SEQ + kc + 16 * half);
        int mv0[8], mv1[8];
#pragma unroll
        for (int r = 0; r < 8; ++r) {
            int qrow = qt * 16 + r + 8 * half;
            const int* mrp = mbase + (size_t)qrow * SEQ + kc;
            mv0[r] = mrp[ln];
            mv1[r] = mrp[16 + ln];
        }
        // ---- prefetch next chunk's K fragments (wrap: tail prefetch harmless)
        v16bf bkn[4];
        loadK((kc + 32) & (SEQ - 1), bkn);

        // ---- scores: two 16x16 n-tiles, each accumulated over Dh in 2 WMMAs
        v8f s0 = {}; s0 = WMMA_BF16(aq0, bk[0], s0); s0 = WMMA_BF16(aq1, bk[1], s0);
        v8f s1 = {}; s1 = WMMA_BF16(aq0, bk[2], s1); s1 = WMMA_BF16(aq1, bk[3], s1);

        // ---- mask + 1/sqrt(Dh) scale, then online softmax
#pragma unroll
        for (int r = 0; r < 8; ++r) {
            float a0 = s0[r] * 0.125f;
            float a1 = s1[r] * 0.125f;
            if (mv0[r] == 0) a0 = -1.25e9f;   // (-1e10) * 0.125
            if (mv1[r] == 0) a1 = -1.25e9f;
            s0[r] = a0; s1[r] = a1;
        }
#pragma unroll
        for (int r = 0; r < 8; ++r) {
            float v = fmaxf(s0[r], s1[r]);          // row max across 32 keys
            v = fmaxf(v, __shfl_xor(v, 1, 32));
            v = fmaxf(v, __shfl_xor(v, 2, 32));
            v = fmaxf(v, __shfl_xor(v, 4, 32));
            v = fmaxf(v, __shfl_xor(v, 8, 32));
            float mn = fmaxf(mrow[r], v);
            float al = __expf(mrow[r] - mn);
            mrow[r]  = mn;
            float p0 = __expf(s0[r] - mn);
            float p1 = __expf(s1[r] - mn);
            s0[r] = p0; s1[r] = p1;
            float rs = p0 + p1;
            rs += __shfl_xor(rs, 1, 32);
            rs += __shfl_xor(rs, 2, 32);
            rs += __shfl_xor(rs, 4, 32);
            rs += __shfl_xor(rs, 8, 32);
            lrow[r] = lrow[r] * al + rs;
#pragma unroll
            for (int t = 0; t < 4; ++t) oc[t][r] *= al;   // rescale running O
        }

        // ---- P (16x32) to LDS row-major, re-read as A fragment
#pragma unroll
        for (int r = 0; r < 8; ++r) {
            int m = r + 8 * half;
            pmine[m * 32 + ln]      = (__bf16)s0[r];
            pmine[m * 32 + 16 + ln] = (__bf16)s1[r];
        }
        v16bf ap = load_a16(pmine + ln * 32 + 16 * half);

        // ---- O += P * V  (V stored transposed: [Dh][S], contiguous over keys)
#pragma unroll
        for (int t = 0; t < 4; ++t) oc[t] = WMMA_BF16(ap, bv[t], oc[t]);

#pragma unroll
        for (int i = 0; i < 4; ++i) bk[i] = bkn[i];
    }

    // ---- normalize and store bf16 into [B,S,D]
#pragma unroll
    for (int t = 0; t < 4; ++t) {
#pragma unroll
        for (int r = 0; r < 8; ++r) {
            int   m = r + 8 * half;
            float v = oc[t][r] / fmaxf(lrow[r], 1e-20f);
            out[((size_t)(b * SEQ + qt * 16 + m)) * DM + h * DH + 16 * t + ln] = (__bf16)v;
        }
    }
}

// ---------------------------------------------------------------------------
extern "C" void kernel_launch(void* const* d_in, const int* in_sizes, int n_in,
                              void* d_out, int out_size, void* d_ws, size_t ws_size,
                              hipStream_t stream) {
    const float* q  = (const float*)d_in[0];
    const float* k  = (const float*)d_in[1];
    const float* v  = (const float*)d_in[2];
    const int*   mk = (const int*)  d_in[3];
    const float* wq = (const float*)d_in[4];
    const float* wk = (const float*)d_in[5];
    const float* wv = (const float*)d_in[6];
    const float* wo = (const float*)d_in[7];
    const float* bo = (const float*)d_in[8];

    // workspace layout (bf16 elements)
    __bf16* base = (__bf16*)d_ws;
    const size_t ACT = (size_t)MTOT * DM;   // 4096*1024
    const size_t WEL = (size_t)DM * DM;     // 1024*1024
    __bf16* xq  = base;
    __bf16* xk  = xq  + ACT;
    __bf16* xv  = xk  + ACT;
    __bf16* wqT = xv  + ACT;
    __bf16* wkT = wqT + WEL;
    __bf16* wvT = wkT + WEL;
    __bf16* woT = wvT + WEL;
    __bf16* qh  = woT + WEL;   // [B,H,S,Dh]
    __bf16* kh  = qh  + ACT;   // [B,H,S,Dh]
    __bf16* vt  = kh  + ACT;   // [B,H,Dh,S]
    __bf16* ao  = vt  + ACT;   // [B,S,D]
    (void)ws_size; (void)in_sizes; (void)n_in; (void)out_size;

    const int n4 = (int)(ACT / 4);
    dim3 blk(256);

    // 1) precision conversion
    cvt_bf16_kernel<<<(n4 + 255) / 256, blk, 0, stream>>>(q, xq, n4);
    cvt_bf16_kernel<<<(n4 + 255) / 256, blk, 0, stream>>>(k, xk, n4);
    cvt_bf16_kernel<<<(n4 + 255) / 256, blk, 0, stream>>>(v, xv, n4);
    cvtT_bf16_kernel<<<(DM * DM + 255) / 256, blk, 0, stream>>>(wq, wqT);
    cvtT_bf16_kernel<<<(DM * DM + 255) / 256, blk, 0, stream>>>(wk, wkT);
    cvtT_bf16_kernel<<<(DM * DM + 255) / 256, blk, 0, stream>>>(wv, wvT);
    cvtT_bf16_kernel<<<(DM * DM + 255) / 256, blk, 0, stream>>>(wo, woT);

    // 2) QKV projections (256 blocks: 64 row-tiles x 4 col-tiles)
    gemm_bf16_kernel<<<256, blk, 0, stream>>>(xq, wqT, nullptr, qh, 0);
    gemm_bf16_kernel<<<256, blk, 0, stream>>>(xk, wkT, nullptr, kh, 0);
    gemm_bf16_kernel<<<256, blk, 0, stream>>>(xv, wvT, nullptr, vt, 1);

    // 3) flash attention: 4096 waves = 512 blocks of 8 waves
    flash_attn_kernel<<<512, blk, 0, stream>>>(qh, kh, vt, mk, ao);

    // 4) output projection + bias (fp32 out)
    gemm_bf16_kernel<<<256, blk, 0, stream>>>(ao, woT, bo, (float*)d_out, 2);
}
